// Hierarchical_HiResPrecipNet_1563368096254
// MI455X (gfx1250) — matrix operand order repaired and checked by reading, not compiled, not run on hardware
//
#include <hip/hip_runtime.h>

// ---------------------------------------------------------------------------
// CDNA5 (gfx1250) hierarchical GATv2 net.
// Dense transforms: LDS-staged bf16 WMMA (v_wmma_f32_16x16x32_bf16, f32 acc).
// Sparse edge softmax/aggregation: gather + f32 global atomics (L2 resident).
// ---------------------------------------------------------------------------

typedef __attribute__((ext_vector_type(16))) __bf16 v16bf;
typedef __attribute__((ext_vector_type(8)))  __bf16 v8bf;
typedef __attribute__((ext_vector_type(2)))  __bf16 v2bf;
typedef __attribute__((ext_vector_type(8)))  float  v8f;

#define N_LOW_   10000
#define N_HIGH_  100000
#define E_H_     80000
#define E_V_     40000
#define E_L2H_   400000
#define E_HH_    800000
#define NEG_SLOPE_ 0.2f

static inline int cdiv_(long a, long b) { return (int)((a + b - 1) / b); }

// ------------------------------ kernels ------------------------------------

__global__ void k_fill(float* __restrict__ p, float v, long n) {
  long i = (long)blockIdx.x * blockDim.x + threadIdx.x;
  if (i < n) p[i] = v;
}

// Y[M,N] = X[M,K] @ W[K,N] + bias[N], optional ReLU.
// Block = 256 threads (8 waves). Block tile = 128 rows x 16 cols.
// K-tiles of 32 staged in LDS as bf16; each wave does one 16x16 WMMA tile.
// OOB rows/cols are address-clamped (their D elements are never stored);
// only K-tail elements are zero-selected. No divergent loads anywhere.
#define APAD 40   // row pitch in bf16 elements (80B: 16B-aligned, bank-spread)
__global__ __launch_bounds__(256)
void k_gemm_bias(const float* __restrict__ X, const float* __restrict__ W,
                 const float* __restrict__ Bias, float* __restrict__ Y,
                 int M, int K, int N, int relu) {
  __shared__ __align__(16) __bf16 Alds[128 * APAD];
  __shared__ __align__(16) __bf16 Blds[16 * APAD];

  const int t      = threadIdx.x;
  const int lane   = t & 31;
  const int wave   = t >> 5;
  const int mtile0 = blockIdx.x * 128;
  const int ntile  = blockIdx.y * 16;
  const int r      = lane & 15;
  const int kaBase = (lane < 16) ? 0 : 8;   // A-matrix 16-bit layout (ISA 7.12.2)
  const int kbBase = (lane < 16) ? 0 : 8;   // B-matrix 16-bit layout

  v8f acc = {0.f, 0.f, 0.f, 0.f, 0.f, 0.f, 0.f, 0.f};

#pragma unroll 2
  for (int k0 = 0; k0 < K; k0 += 32) {
    __syncthreads();
    // ---- stage A tile (128x32) : 2 consecutive-k elems per thread x 8 ----
#pragma unroll
    for (int i = 0; i < 8; ++i) {
      int idx = t * 2 + i * 512;            // 0..4094
      int rr  = idx >> 5;                   // 0..127
      int kk  = idx & 31;                   // even
      int gr  = min(mtile0 + rr, M - 1);    // clamp: safe, row-private
      int gk0 = k0 + kk, gk1 = gk0 + 1;
      float v0 = X[(long)gr * K + min(gk0, K - 1)];
      float v1 = X[(long)gr * K + min(gk1, K - 1)];
      v2bf p;
      p[0] = (gk0 < K) ? (__bf16)v0 : (__bf16)0.f;
      p[1] = (gk1 < K) ? (__bf16)v1 : (__bf16)0.f;
      *(v2bf*)(&Alds[rr * APAD + kk]) = p;
    }
    // ---- stage B tile (32x16) transposed to [n][k] : 2 elems per thread ----
    {
      int nn  = t >> 4;                     // 0..15
      int kk  = (t & 15) * 2;               // 0..30
      int gn  = min(ntile + nn, N - 1);     // clamp: safe, col-private
      int gk0 = k0 + kk, gk1 = gk0 + 1;
      float v0 = W[(long)min(gk0, K - 1) * N + gn];
      float v1 = W[(long)min(gk1, K - 1) * N + gn];
      v2bf p;
      p[0] = (gk0 < K) ? (__bf16)v0 : (__bf16)0.f;
      p[1] = (gk1 < K) ? (__bf16)v1 : (__bf16)0.f;
      *(v2bf*)(&Blds[nn * APAD + kk]) = p;
    }
    __syncthreads();

    // ---- fetch WMMA operands from LDS (aligned b128 reads) ----
    const __bf16* ar = &Alds[(wave * 16 + r) * APAD];
    v8bf a0 = *(const v8bf*)(ar + kaBase);
    v8bf a1 = *(const v8bf*)(ar + 16 + kaBase);
    v16bf a = __builtin_shufflevector(a0, a1, 0, 1, 2, 3, 4, 5, 6, 7,
                                      8, 9, 10, 11, 12, 13, 14, 15);
    const __bf16* br = &Blds[r * APAD];
    v8bf b0 = *(const v8bf*)(br + kbBase);
    v8bf b1 = *(const v8bf*)(br + 16 + kbBase);
    v16bf b = __builtin_shufflevector(b0, b1, 0, 1, 2, 3, 4, 5, 6, 7,
                                      8, 9, 10, 11, 12, 13, 14, 15);

    acc = __builtin_amdgcn_wmma_f32_16x16x32_bf16(
        /*neg_a=*/false, a, /*neg_b=*/false, b,
        /*c_mod=*/(short)0, acc, /*reuse_a=*/false, /*reuse_b=*/false);
  }

  // ---- epilogue: bias + optional relu ----
  // C/D layout: VGPR j -> row = orow0 + j, col = ntile + (lane&15)
  const int col   = ntile + r;
  const int orow0 = mtile0 + wave * 16 + ((lane < 16) ? 0 : 8);
  float bias = Bias[min(col, N - 1)];
  const bool fullTile = (mtile0 + 128 <= M) && (ntile + 16 <= N);
  if (fullTile) {                           // block-uniform fast path: no guards
    float* yp = Y + (long)orow0 * N + col;
#pragma unroll
    for (int j = 0; j < 8; ++j) {
      float v = acc[j] + bias;
      if (relu) v = fmaxf(v, 0.f);
      yp[(long)j * N] = v;
    }
  } else {                                  // M- or N-tail edge blocks: guarded stores
#pragma unroll
    for (int j = 0; j < 8; ++j) {
      int orow = orow0 + j;
      if (orow < M && col < N) {
        float v = acc[j] + bias;
        if (relu) v = fmaxf(v, 0.f);
        Y[(long)orow * N + col] = v;
      }
    }
  }
}

// per-(edge,head) GATv2 attention logit
__global__ void k_edge_logits(const float* __restrict__ xl, const float* __restrict__ xr,
                              const int* __restrict__ src, const int* __restrict__ dst,
                              const float* __restrict__ att, float* __restrict__ logit,
                              int E, int Etot, int H, int C) {
  long t = (long)blockIdx.x * blockDim.x + threadIdx.x;
  if (t >= (long)Etot * H) return;
  int e = (int)(t / H), h = (int)(t % H);
  int s = (e < E) ? src[e] : (e - E);     // tail = self loops
  int d = (e < E) ? dst[e] : (e - E);
  long sb = (long)s * H * C + (long)h * C;
  long db = (long)d * H * C + (long)h * C;
  const float* ap = att + (long)h * C;
  float acc = 0.f;
  for (int c = 0; c < C; ++c) {
    float v = xl[sb + c] + xr[db + c];
    v = (v > 0.f) ? v : NEG_SLOPE_ * v;
    acc += v * ap[c];
  }
  logit[(long)e * H + h] = acc;
}

__device__ __forceinline__ void atomicMaxF(float* a, float v) {
  if (v >= 0.f) atomicMax((int*)a, __float_as_int(v));
  else          atomicMin((unsigned int*)a, __float_as_uint(v));
}

__global__ void k_seg_max(const float* __restrict__ logit, const int* __restrict__ dst,
                          float* __restrict__ mx, int E, int Etot, int H) {
  long t = (long)blockIdx.x * blockDim.x + threadIdx.x;
  if (t >= (long)Etot * H) return;
  int e = (int)(t / H), h = (int)(t % H);
  int d = (e < E) ? dst[e] : (e - E);
  atomicMaxF(&mx[(long)d * H + h], logit[(long)e * H + h]);
}

// in-place: logit -> exp(logit - max); accumulate denominator and degree
__global__ void k_edge_exp(float* __restrict__ logit, const float* __restrict__ mx,
                           float* __restrict__ den, float* __restrict__ deg,
                           const int* __restrict__ dst, int E, int Etot, int H) {
  long t = (long)blockIdx.x * blockDim.x + threadIdx.x;
  if (t >= (long)Etot * H) return;
  int e = (int)(t / H), h = (int)(t % H);
  int d = (e < E) ? dst[e] : (e - E);
  long idx = (long)e * H + h;
  float w = __expf(logit[idx] - mx[(long)d * H + h]);
  logit[idx] = w;
  atomicAdd(&den[(long)d * H + h], w);
  if (h == 0) atomicAdd(&deg[d], 1.0f);
}

__global__ void k_aggr(const float* __restrict__ ex, const float* __restrict__ den,
                       const float* __restrict__ xl,
                       const int* __restrict__ src, const int* __restrict__ dst,
                       float* __restrict__ out, int E, int Etot, int H, int C) {
  long t = (long)blockIdx.x * blockDim.x + threadIdx.x;
  if (t >= (long)Etot * H) return;
  int e = (int)(t / H), h = (int)(t % H);
  int s = (e < E) ? src[e] : (e - E);
  int d = (e < E) ? dst[e] : (e - E);
  float a = ex[(long)e * H + h] / (den[(long)d * H + h] + 1e-16f);
  long sb = (long)s * H * C + (long)h * C;
  long ob = (long)d * H * C + (long)h * C;
  for (int c = 0; c < C; ++c) atomicAdd(&out[ob + c], a * xl[sb + c]);
}

__global__ void k_gat_final(float* __restrict__ out, const float* __restrict__ deg,
                            const float* __restrict__ bout, int Nn, int HC, int relu) {
  long t = (long)blockIdx.x * blockDim.x + threadIdx.x;
  if (t >= (long)Nn * HC) return;
  int n = (int)(t / HC), j = (int)(t % HC);
  float v = out[t] / fmaxf(deg[n], 1.f) + bout[j];
  if (relu) v = fmaxf(v, 0.f);
  out[t] = v;
}

// one block per column: mean + biased variance
__global__ __launch_bounds__(256)
void k_bn_stats(const float* __restrict__ x, int Nn, int D,
                float* __restrict__ mu, float* __restrict__ var) {
  __shared__ float ss[256], sq[256];
  int d = blockIdx.x;
  float s = 0.f, q = 0.f;
  for (int i = threadIdx.x; i < Nn; i += blockDim.x) {
    float v = x[(long)i * D + d];
    s += v; q += v * v;
  }
  ss[threadIdx.x] = s; sq[threadIdx.x] = q;
  __syncthreads();
  for (int st = 128; st > 0; st >>= 1) {
    if ((int)threadIdx.x < st) {
      ss[threadIdx.x] += ss[threadIdx.x + st];
      sq[threadIdx.x] += sq[threadIdx.x + st];
    }
    __syncthreads();
  }
  if (threadIdx.x == 0) {
    float m = ss[0] / Nn;
    mu[d] = m;
    var[d] = sq[0] / Nn - m * m;
  }
}

__global__ void k_bn_apply(float* __restrict__ x, const float* __restrict__ mu,
                           const float* __restrict__ var, const float* __restrict__ g,
                           const float* __restrict__ b, int Nn, int D) {
  long t = (long)blockIdx.x * blockDim.x + threadIdx.x;
  if (t >= (long)Nn * D) return;
  int d = (int)(t % D);
  float v = (x[t] - mu[d]) * __frsqrt_rn(var[d] + 1e-5f) * g[d] + b[d];
  x[t] = fmaxf(v, 0.f);                   // BN always followed by ReLU here
}

__global__ void k_concat(const float* __restrict__ z, const float* __restrict__ enc,
                         float* __restrict__ out, int Nn) {
  int i = blockIdx.x * blockDim.x + threadIdx.x;
  if (i >= Nn) return;
  out[(long)i * 65] = z[i];
  for (int c = 0; c < 64; ++c) out[(long)i * 65 + 1 + c] = enc[(long)i * 64 + c];
}

// ------------------------------ host side ----------------------------------

struct GatP { const float *Wl, *bl, *Wr, *br, *att, *bout; };

static void gemm_(hipStream_t s, const float* X, const float* W, const float* B,
                  float* Y, int M, int K, int N, int relu) {
  dim3 g(cdiv_(M, 128), cdiv_(N, 16));
  k_gemm_bias<<<g, 256, 0, s>>>(X, W, B, Y, M, K, N, relu);
}

static void run_gat(hipStream_t s,
                    const float* xs, int ns, int ds,
                    const float* xd, int nd, int dd,
                    const int* ei, int E, bool loops,
                    const GatP& p, int H, int C,
                    float* out, int relu,
                    float* xl, float* xr, float* logit,
                    float* mx, float* den, float* deg) {
  const int HC = H * C;
  const int Etot = E + (loops ? nd : 0);
  const int* src = ei;
  const int* dst = ei + E;
  gemm_(s, xs, p.Wl, p.bl, xl, ns, ds, HC, 0);
  gemm_(s, xd, p.Wr, p.br, xr, nd, dd, HC, 0);
  k_fill<<<cdiv_((long)nd * H, 256), 256, 0, s>>>(mx, -3.0e38f, (long)nd * H);
  k_fill<<<cdiv_((long)nd * H, 256), 256, 0, s>>>(den, 0.f, (long)nd * H);
  k_fill<<<cdiv_(nd, 256), 256, 0, s>>>(deg, 0.f, nd);
  k_fill<<<cdiv_((long)nd * HC, 256), 256, 0, s>>>(out, 0.f, (long)nd * HC);
  long t = (long)Etot * H;
  k_edge_logits<<<cdiv_(t, 256), 256, 0, s>>>(xl, xr, src, dst, p.att, logit, E, Etot, H, C);
  k_seg_max<<<cdiv_(t, 256), 256, 0, s>>>(logit, dst, mx, E, Etot, H);
  k_edge_exp<<<cdiv_(t, 256), 256, 0, s>>>(logit, mx, den, deg, dst, E, Etot, H);
  k_aggr<<<cdiv_(t, 256), 256, 0, s>>>(logit, den, xl, src, dst, out, E, Etot, H, C);
  k_gat_final<<<cdiv_((long)nd * HC, 256), 256, 0, s>>>(out, deg, p.bout, nd, HC, relu);
}

static inline float* balloc_(char*& p, size_t bytes) {
  float* r = (float*)p;
  p += (bytes + 255) & ~(size_t)255;
  return r;
}

extern "C" void kernel_launch(void* const* d_in, const int* in_sizes, int n_in,
                              void* d_out, int out_size, void* d_ws, size_t ws_size,
                              hipStream_t stream) {
  (void)in_sizes; (void)n_in; (void)out_size; (void)ws_size;

  // ---- unpack inputs (setup_inputs() flat order) ----
  const float* x_low[5];
  for (int i = 0; i < 5; ++i) x_low[i] = (const float*)d_in[i];
  const float* x_high = (const float*)d_in[5];
  const float* z_std  = (const float*)d_in[6];
  auto GP = [&](int b) {
    GatP g;
    g.Wl  = (const float*)d_in[b + 0]; g.bl   = (const float*)d_in[b + 1];
    g.Wr  = (const float*)d_in[b + 2]; g.br   = (const float*)d_in[b + 3];
    g.att = (const float*)d_in[b + 4]; g.bout = (const float*)d_in[b + 5];
    return g;
  };
  GatP horiz[3] = { GP(7), GP(13), GP(19) };
  GatP vert  = GP(25);
  GatP down  = GP(31);
  GatP proc1 = GP(37);
  const float* bn1g = (const float*)d_in[43]; const float* bn1b = (const float*)d_in[44];
  GatP proc2 = GP(45);
  const float* bn2g = (const float*)d_in[51]; const float* bn2b = (const float*)d_in[52];
  GatP proc3 = GP(53);
  const float* bn3g = (const float*)d_in[59]; const float* bn3b = (const float*)d_in[60];
  const float* W1 = (const float*)d_in[61]; const float* b1 = (const float*)d_in[62];
  const float* W2 = (const float*)d_in[63]; const float* b2 = (const float*)d_in[64];
  const float* W3 = (const float*)d_in[65]; const float* b3 = (const float*)d_in[66];
  const int* ei_h[5]; for (int i = 0; i < 5; ++i) ei_h[i] = (const int*)d_in[67 + i];
  const int* ei_v[4]; for (int i = 0; i < 4; ++i) ei_v[i] = (const int*)d_in[72 + i];
  const int* ei_l2h  = (const int*)d_in[76];
  const int* ei_high = (const int*)d_in[77];

  // ---- workspace layout (bump allocator) ----
  char* wp = (char*)d_ws;
  float* xl    = balloc_(wp, (size_t)N_HIGH_ * 128 * 4);
  float* xr    = balloc_(wp, (size_t)N_HIGH_ * 128 * 4);
  float* logit = balloc_(wp, (size_t)(E_HH_ + N_HIGH_) * 2 * 4);
  float* mx    = balloc_(wp, (size_t)N_HIGH_ * 2 * 4);
  float* den   = balloc_(wp, (size_t)N_HIGH_ * 2 * 4);
  float* deg   = balloc_(wp, (size_t)N_HIGH_ * 4);
  float* e65   = balloc_(wp, (size_t)N_HIGH_ * 65 * 4);
  float* hA    = balloc_(wp, (size_t)N_HIGH_ * 128 * 4);
  float* hB    = balloc_(wp, (size_t)N_HIGH_ * 128 * 4);
  float* L[5];
  for (int i = 0; i < 5; ++i) L[i] = balloc_(wp, (size_t)N_LOW_ * 25 * 4);
  float* S  = balloc_(wp, (size_t)N_LOW_ * 25 * 4);
  float* mu = balloc_(wp, 128 * 4);
  float* vr = balloc_(wp, 128 * 4);

  // ---- low-res horizontal stacks (3 shared layers per level) ----
  for (int i = 0; i < 5; ++i) {
    run_gat(stream, x_low[i], N_LOW_, 25, x_low[i], N_LOW_, 25, ei_h[i], E_H_, false,
            horiz[0], 1, 25, L[i], 1, xl, xr, logit, mx, den, deg);
    run_gat(stream, L[i], N_LOW_, 25, L[i], N_LOW_, 25, ei_h[i], E_H_, false,
            horiz[1], 1, 25, S, 1, xl, xr, logit, mx, den, deg);
    run_gat(stream, S, N_LOW_, 25, S, N_LOW_, 25, ei_h[i], E_H_, false,
            horiz[2], 1, 25, L[i], 1, xl, xr, logit, mx, den, deg);
  }

  // ---- vertical chain (shared params), 200->500->700->850->1000 ----
  float* h[5] = { L[0], L[1], L[2], L[3], L[4] };
  for (int v = 0; v < 4; ++v) {
    run_gat(stream, h[v], N_LOW_, 25, h[v + 1], N_LOW_, 25, ei_v[v], E_V_, false,
            vert, 1, 25, S, 1, xl, xr, logit, mx, den, deg);
    float* t = h[v + 1]; h[v + 1] = S; S = t;
  }

  // ---- downscale low->high (no relu), concat with z_std -> 65ch ----
  run_gat(stream, h[4], N_LOW_, 25, x_high, N_HIGH_, 1, ei_l2h, E_L2H_, false,
          down, 1, 64, hA, 0, xl, xr, logit, mx, den, deg);
  k_concat<<<cdiv_(N_HIGH_, 256), 256, 0, stream>>>(z_std, hA, e65, N_HIGH_);

  // ---- processor GATs on high-res graph (self loops, BN+ReLU) ----
  auto bn = [&](float* x, int D, const float* g, const float* b) {
    k_bn_stats<<<D, 256, 0, stream>>>(x, N_HIGH_, D, mu, vr);
    k_bn_apply<<<cdiv_((long)N_HIGH_ * D, 256), 256, 0, stream>>>(x, mu, vr, g, b, N_HIGH_, D);
  };
  run_gat(stream, e65, N_HIGH_, 65, e65, N_HIGH_, 65, ei_high, E_HH_, true,
          proc1, 2, 64, hA, 0, xl, xr, logit, mx, den, deg);
  bn(hA, 128, bn1g, bn1b);
  run_gat(stream, hA, N_HIGH_, 128, hA, N_HIGH_, 128, ei_high, E_HH_, true,
          proc2, 2, 64, hB, 0, xl, xr, logit, mx, den, deg);
  bn(hB, 128, bn2g, bn2b);
  run_gat(stream, hB, N_HIGH_, 128, hB, N_HIGH_, 128, ei_high, E_HH_, true,
          proc3, 1, 64, hA, 0, xl, xr, logit, mx, den, deg);
  bn(hA, 64, bn3g, bn3b);

  // ---- predictor MLP 64 -> 64 -> 32 -> 1 ----
  gemm_(stream, hA, W1, b1, hB, N_HIGH_, 64, 64, 1);
  gemm_(stream, hB, W2, b2, hA, N_HIGH_, 64, 32, 1);
  gemm_(stream, hA, W3, b3, (float*)d_out, N_HIGH_, 32, 1, 0);
}
